// NonLocalBlock2D_34883724378664
// MI455X (gfx1250) — matrix operand order repaired and checked by reading, not compiled, not run on hardware
//
#include <hip/hip_runtime.h>
#include <hip/hip_bf16.h>

typedef __attribute__((ext_vector_type(16))) _Float16 v16h;
typedef __attribute__((ext_vector_type(8)))  _Float16 v8h;
typedef __attribute__((ext_vector_type(4)))  _Float16 v4h;
typedef __attribute__((ext_vector_type(8)))  float    v8f;

#define WMMA_F32_F16(a, b, c) \
  __builtin_amdgcn_wmma_f32_16x16x32_f16(false, (a), false, (b), (short)0, (c), false, false)

// ---------------------------------------------------------------------------
// Fragment loaders per CDNA5 VGPR layouts (CDNA5 ISA §7.12.2)
// A (16x32 f16): lane L -> row M=L%16; a[0..7]=row[half*8 .. +7],
//                a[8..15]=row[16+half*8 .. +7], half = L/16.
// B (32x16 f16): lane L -> col N=L%16; b[0..7]=K[half*8 .. +7], b[8..15]=K[16+half*8 .. +7].
// C/D (16x16 f32): VGPR r, lane l -> M = r + 8*(l/16), N = l%16.
// ---------------------------------------------------------------------------
static __device__ __forceinline__ v16h load_a_frag(const _Float16* tile, int ld, int lane) {
  const _Float16* row = tile + (size_t)(lane & 15) * ld + ((lane >> 4) & 1) * 8;
  v8h lo = *(const v8h*)(row);
  v8h hi = *(const v8h*)(row + 16);
  v16h a;
#pragma unroll
  for (int j = 0; j < 8; ++j) { a[j] = lo[j]; a[j + 8] = hi[j]; }
  return a;
}

static __device__ __forceinline__ v16h load_b_frag(const _Float16* tile, int ld, int lane) {
  const _Float16* colp = tile + (lane & 15) + (size_t)(((lane >> 4) & 1) * 8) * ld;
  v16h b;
#pragma unroll
  for (int j = 0; j < 8; ++j) { b[j] = colp[(size_t)j * ld]; b[j + 8] = colp[(size_t)(16 + j) * ld]; }
  return b;
}

// ---------------------------------------------------------------------------
// Weight prep: f32 -> f16, wq/wv transposed (for (N,C)-layout outputs),
// wk straight (for (C,N)-layout output).
// ---------------------------------------------------------------------------
__global__ __launch_bounds__(256) void prep_weights_kernel(
    const float* __restrict__ wq, const float* __restrict__ wk, const float* __restrict__ wv,
    _Float16* __restrict__ wqT, _Float16* __restrict__ wk16, _Float16* __restrict__ wvT) {
  const int co = blockIdx.x;
  for (int ci = threadIdx.x; ci < 512; ci += 256) {
    wqT[(size_t)ci * 512 + co]  = (_Float16)wq[(size_t)co * 512 + ci];
    wk16[(size_t)co * 512 + ci] = (_Float16)wk[(size_t)co * 512 + ci];
    wvT[(size_t)ci * 512 + co]  = (_Float16)wv[(size_t)co * 512 + ci];
  }
}

// ---------------------------------------------------------------------------
// InstanceNorm (affine=False): per (b,c) over 4096 pixels, biased variance.
// Emits normalized activations in both (C,N) and (N,C) f16 layouts.
// ---------------------------------------------------------------------------
__global__ __launch_bounds__(256) void instnorm_kernel(
    const float* __restrict__ x, _Float16* __restrict__ h_cn, _Float16* __restrict__ h_nc) {
  const int bc = blockIdx.x;            // b*512 + c
  const int b = bc >> 9, c = bc & 511;
  const float4* row4 = (const float4*)(x + (size_t)bc * 4096);
  const int tid = threadIdx.x;
  float s = 0.f, s2 = 0.f;
  for (int i = tid; i < 1024; i += 256) {
    float4 v = row4[i];
    s  += v.x + v.y + v.z + v.w;
    s2 += v.x * v.x + v.y * v.y + v.z * v.z + v.w * v.w;
  }
  __shared__ float rs[256], rs2[256];
  rs[tid] = s; rs2[tid] = s2;
  __syncthreads();
  for (int off = 128; off > 0; off >>= 1) {
    if (tid < off) { rs[tid] += rs[tid + off]; rs2[tid] += rs2[tid + off]; }
    __syncthreads();
  }
  const float mean = rs[0] * (1.f / 4096.f);
  const float var  = rs2[0] * (1.f / 4096.f) - mean * mean;
  const float inv  = rsqrtf(var + 1e-5f);
  for (int i = tid; i < 1024; i += 256) {
    float4 v = row4[i];
    v4h h;
    h[0] = (_Float16)((v.x - mean) * inv);
    h[1] = (_Float16)((v.y - mean) * inv);
    h[2] = (_Float16)((v.z - mean) * inv);
    h[3] = (_Float16)((v.w - mean) * inv);
    *(v4h*)(h_cn + (size_t)bc * 4096 + i * 4) = h;
    _Float16* dst = h_nc + ((size_t)b * 4096 + i * 4) * 512 + c;
    dst[0] = h[0]; dst[512] = h[1]; dst[1024] = h[2]; dst[1536] = h[3];
  }
}

// ---------------------------------------------------------------------------
// f16 WMMA GEMM, K fixed at 512: D(M,N) = A(M,512) x B(512,N) + bias.
// 4 waves/block; wave owns a 16x64 output tile. Manual register double-buffer
// of fragments so VMEM for step k+1 overlaps the 4 WMMAs of step k.
// bias_per_row: 1 -> bias[m] (K projection), 0 -> bias[n] (Q/V projections).
// ---------------------------------------------------------------------------
__global__ __launch_bounds__(128) void gemm_f16_kernel(
    const _Float16* __restrict__ A, long long strideA,
    const _Float16* __restrict__ B, long long strideB,
    _Float16* __restrict__ D, long long strideD,
    const float* __restrict__ bias, int bias_per_row,
    int lda, int ldb, int ldd) {
  constexpr int KDIM = 512;
  const int lane = threadIdx.x & 31;
  const int wave = threadIdx.x >> 5;
  const int m0 = blockIdx.x * 64 + wave * 16;
  const int n0 = blockIdx.y * 64;
  A += (size_t)blockIdx.z * strideA;
  B += (size_t)blockIdx.z * strideB;
  D += (size_t)blockIdx.z * strideD;
  const _Float16* Arow = A + (size_t)m0 * lda;

  v8f z = {};
  v8f acc[4];
#pragma unroll
  for (int t = 0; t < 4; ++t) acc[t] = z;

  v16h a_cur = load_a_frag(Arow, lda, lane);
  v16h bc[4];
#pragma unroll
  for (int t = 0; t < 4; ++t) bc[t] = load_b_frag(B + n0 + t * 16, ldb, lane);

#pragma unroll
  for (int kk = 0; kk < KDIM; kk += 32) {
    v16h a_nxt = a_cur;
    v16h bn[4];
    if (kk + 32 < KDIM) {           // resolved at compile time (full unroll)
      a_nxt = load_a_frag(Arow + kk + 32, lda, lane);
#pragma unroll
      for (int t = 0; t < 4; ++t)
        bn[t] = load_b_frag(B + (size_t)(kk + 32) * ldb + n0 + t * 16, ldb, lane);
    } else {
#pragma unroll
      for (int t = 0; t < 4; ++t) bn[t] = bc[t];
    }
#pragma unroll
    for (int t = 0; t < 4; ++t) acc[t] = WMMA_F32_F16(a_cur, bc[t], acc[t]);
    a_cur = a_nxt;
#pragma unroll
    for (int t = 0; t < 4; ++t) bc[t] = bn[t];
  }

  const int rowoff = 8 * (lane >> 4);
  const int col = lane & 15;
#pragma unroll
  for (int t = 0; t < 4; ++t)
#pragma unroll
    for (int r = 0; r < 8; ++r) {
      const int m = m0 + r + rowoff;
      const int n = n0 + t * 16 + col;
      const float v = acc[t][r] + (bias_per_row ? bias[m] : bias[n]);
      D[(size_t)m * ldd + n] = (_Float16)v;
    }
}

// ---------------------------------------------------------------------------
// Flash attention. Qt:(B,N,C) f16, Kc:(B,C,N) f16, Vt:(B,N,C) f16.
// 8 waves/block; each wave owns 16 queries with the full C=512 output resident
// (32 f32 tiles = 256 VGPRs) plus all 16 Q A-fragments (128 VGPRs).
// K/V 32-key slices are staged into LDS via CDNA5 async-to-LDS copies
// (global_load_async_to_lds_b128 + s_wait_asynccnt), ping-pong buffered so the
// next slice's DMA overlaps this slice's 64 WMMAs. Rows are padded (K: 40 f16,
// V: 520 f16) to reduce LDS bank conflicts while keeping 16B alignment.
// ---------------------------------------------------------------------------
#define K_ROW 40     // f16 per staged K row (32 payload + 8 pad)
#define V_ROW 520    // f16 per staged V row (512 payload + 8 pad)

__global__ __launch_bounds__(256, 1) void attn_kernel(
    const _Float16* __restrict__ Qt, const _Float16* __restrict__ Kc,
    const _Float16* __restrict__ Vt, float* __restrict__ o_ws) {
  const int tid = threadIdx.x;
  const int lane = tid & 31;
  const int wave = tid >> 5;
  const int b = blockIdx.y;
  const int q0 = (blockIdx.x * 8 + wave) * 16;
  const _Float16* Qb = Qt + (size_t)b * 4096 * 512;
  const _Float16* Kb = Kc + (size_t)b * 512 * 4096;
  const _Float16* Vb = Vt + (size_t)b * 4096 * 512;

  __shared__ __align__(32) _Float16 k_st[2][512 * K_ROW];  // (c, 32 keys) padded
  __shared__ __align__(32) _Float16 v_st[2][32 * V_ROW];   // (key, 512 ch) padded
  __shared__ __align__(32) _Float16 p_lds[8][512];         // 16x32 P tile per wave

  // Async global->LDS fill of one K/V slice (64 KB payload, 16 b128 per thread).
  auto issue_fill = [&](int buf, int k0) {
    const unsigned kbase = (unsigned)(unsigned long long)(&k_st[buf][0]);
    const unsigned vbase = (unsigned)(unsigned long long)(&v_st[buf][0]);
#pragma unroll
    for (int i = 0; i < 8; ++i) {
      const int id = tid + i * 256;
      // K slice: 512 rows x 4 chunks of 16B
      const int kr = id >> 2, ko = id & 3;
      const unsigned lk = kbase + (unsigned)(kr * (K_ROW * 2) + ko * 16);
      const unsigned long long gk =
          (unsigned long long)(Kb + (size_t)kr * 4096 + k0 + ko * 8);
      asm volatile("global_load_async_to_lds_b128 %0, %1, off"
                   :: "v"(lk), "v"(gk) : "memory");
      // V slice: 32 rows x 64 chunks of 16B
      const int vr = id >> 6, vo = id & 63;
      const unsigned lv = vbase + (unsigned)(vr * (V_ROW * 2) + vo * 16);
      const unsigned long long gv =
          (unsigned long long)(Vb + (size_t)(k0 + vr) * 512 + vo * 8);
      asm volatile("global_load_async_to_lds_b128 %0, %1, off"
                   :: "v"(lv), "v"(gv) : "memory");
    }
  };

  issue_fill(0, 0);

  // Q A-fragments for the whole 16x512 query tile stay in registers.
  v16h qa[16];
#pragma unroll
  for (int cc = 0; cc < 16; ++cc)
    qa[cc] = load_a_frag(Qb + (size_t)q0 * 512 + cc * 32, 512, lane);

  v8f z = {};
  v8f o[32];
#pragma unroll
  for (int t = 0; t < 32; ++t) o[t] = z;
  float m_run[8], l_run[8];
#pragma unroll
  for (int r = 0; r < 8; ++r) { m_run[r] = -1e30f; l_run[r] = 0.f; }

  _Float16* pw = &p_lds[wave][0];
  const int rowoff = 8 * (lane >> 4);
  const int col = lane & 15;
  const float scale = 0.04419417382415922f;        // 512^-0.5

  asm volatile("s_wait_asynccnt 0x0" ::: "memory");
  __syncthreads();

  int cur = 0;
  for (int k0 = 0; k0 < 4096; k0 += 32) {
    if (k0 + 32 < 4096) issue_fill(cur ^ 1, k0 + 32);   // overlap DMA with WMMA
    const _Float16* ks = &k_st[cur][0];
    const _Float16* vs = &v_st[cur][0];

    // ---- S = Q^T K for 32 keys (two 16x16 tiles), contraction over C=512 ----
    v8f s0 = z, s1 = z;
#pragma unroll
    for (int cc = 0; cc < 16; ++cc) {
      const _Float16* krow = ks + (size_t)(cc * 32) * K_ROW;
      v16h b0 = load_b_frag(krow, K_ROW, lane);       // keys 0..15
      v16h b1 = load_b_frag(krow + 16, K_ROW, lane);  // keys 16..31
      s0 = WMMA_F32_F16(qa[cc], b0, s0);
      s1 = WMMA_F32_F16(qa[cc], b1, s1);
    }
    // ---- online softmax: row max/sum across the 16-lane N-group ----
    float mx[8];
#pragma unroll
    for (int r = 0; r < 8; ++r) {
      s0[r] *= scale; s1[r] *= scale;
      mx[r] = fmaxf(s0[r], s1[r]);
    }
#pragma unroll
    for (int off = 1; off < 16; off <<= 1)
#pragma unroll
      for (int r = 0; r < 8; ++r)
        mx[r] = fmaxf(mx[r], __shfl_xor(mx[r], off, 32));
    float alpha[8], psum[8];
#pragma unroll
    for (int r = 0; r < 8; ++r) {
      const float mnew = fmaxf(m_run[r], mx[r]);
      alpha[r] = __expf(m_run[r] - mnew);
      m_run[r] = mnew;
      const float p0 = __expf(s0[r] - mnew);
      const float p1 = __expf(s1[r] - mnew);
      psum[r] = p0 + p1;
      const int q = r + rowoff;
      pw[q * 32 + col]      = (_Float16)p0;   // D-layout -> row-major P in LDS
      pw[q * 32 + 16 + col] = (_Float16)p1;
    }
#pragma unroll
    for (int off = 1; off < 16; off <<= 1)
#pragma unroll
      for (int r = 0; r < 8; ++r)
        psum[r] += __shfl_xor(psum[r], off, 32);
#pragma unroll
    for (int r = 0; r < 8; ++r) l_run[r] = l_run[r] * alpha[r] + psum[r];
#pragma unroll
    for (int t = 0; t < 32; ++t)
#pragma unroll
      for (int r = 0; r < 8; ++r) o[t][r] *= alpha[r];

    // ---- O += P * V^T : 32 channel tiles, contraction over the 32 keys ----
    v16h pa = load_a_frag(pw, 32, lane);      // wave-private; DScnt-ordered
#pragma unroll
    for (int t = 0; t < 32; ++t) {
      v16h vb = load_b_frag(vs + t * 16, V_ROW, lane);
      o[t] = WMMA_F32_F16(pa, vb, o[t]);
    }

    asm volatile("s_wait_asynccnt 0x0" ::: "memory");  // next slice landed
    __syncthreads();   // everyone done reading `cur`; uniform trip count
    cur ^= 1;
  }

  // ---- finalize: O / l, store (b, n, c) f32 ----
  float rl[8];
#pragma unroll
  for (int r = 0; r < 8; ++r) rl[r] = 1.f / l_run[r];
#pragma unroll
  for (int t = 0; t < 32; ++t)
#pragma unroll
    for (int r = 0; r < 8; ++r) {
      const int q = q0 + r + rowoff;
      const int c = t * 16 + col;
      o_ws[((size_t)b * 4096 + q) * 512 + c] = o[t][r] * rl[r];
    }
}

// ---------------------------------------------------------------------------
// Residual epilogue: out[b,c,n] = x[b,c,n] + O[b,n,c]
// ---------------------------------------------------------------------------
__global__ __launch_bounds__(256) void epilogue_kernel(
    const float* __restrict__ x, const float* __restrict__ o_ws, float* __restrict__ out) {
  const int bc = blockIdx.x;
  const int b = bc >> 9, c = bc & 511;
  for (int n = threadIdx.x; n < 4096; n += 256)
    out[(size_t)bc * 4096 + n] =
        x[(size_t)bc * 4096 + n] + o_ws[((size_t)b * 4096 + n) * 512 + c];
}

extern "C" void kernel_launch(void* const* d_in, const int* in_sizes, int n_in,
                              void* d_out, int out_size, void* d_ws, size_t ws_size,
                              hipStream_t stream) {
  const float* x  = (const float*)d_in[0];
  const float* wq = (const float*)d_in[1];
  const float* bq = (const float*)d_in[2];
  const float* wk = (const float*)d_in[3];
  const float* bk = (const float*)d_in[4];
  const float* wv = (const float*)d_in[5];
  const float* bv = (const float*)d_in[6];
  float* out = (float*)d_out;

  char* ws = (char*)d_ws;
  const size_t MB = 1024ull * 1024ull;
  _Float16* h_cn = (_Float16*)(ws + 0);         // (B,C,N) f16, 16 MB
  _Float16* h_nc = (_Float16*)(ws + 16 * MB);   // (B,N,C) f16, 16 MB
  float*    o_ws = (float*)(ws + 0);            // (B,N,C) f32, aliases h buffers (dead by then)
  _Float16* Qt   = (_Float16*)(ws + 32 * MB);   // (B,N,C) f16
  _Float16* Kc   = (_Float16*)(ws + 48 * MB);   // (B,C,N) f16
  _Float16* Vt   = (_Float16*)(ws + 64 * MB);   // (B,N,C) f16
  _Float16* wqT  = (_Float16*)(ws + 80 * MB);
  _Float16* wk16 = (_Float16*)(ws + 80 * MB + 512 * 1024);
  _Float16* wvT  = (_Float16*)(ws + 81 * MB);

  prep_weights_kernel<<<512, 256, 0, stream>>>(wq, wk, wv, wqT, wk16, wvT);
  instnorm_kernel<<<2048, 256, 0, stream>>>(x, h_cn, h_nc);
  // Qt = h_nc(4096x512) x wqT(512x512) + bq[col]
  gemm_f16_kernel<<<dim3(64, 8, 4), 128, 0, stream>>>(
      h_nc, 4096ll * 512, wqT, 0, Qt, 4096ll * 512, bq, 0, 512, 512, 512);
  // Kc = wk16(512x512) x h_cn(512x4096) + bk[row]
  gemm_f16_kernel<<<dim3(8, 64, 4), 128, 0, stream>>>(
      wk16, 0, h_cn, 512ll * 4096, Kc, 512ll * 4096, bk, 1, 512, 4096, 4096);
  // Vt = h_nc(4096x512) x wvT(512x512) + bv[col]
  gemm_f16_kernel<<<dim3(64, 8, 4), 128, 0, stream>>>(
      h_nc, 4096ll * 512, wvT, 0, Vt, 4096ll * 512, bv, 0, 512, 512, 512);
  attn_kernel<<<dim3(32, 4), 256, 0, stream>>>(Qt, Kc, Vt, o_ws);
  epilogue_kernel<<<2048, 256, 0, stream>>>(x, o_ws, out);
}
